// HRNNModel_79302276153838
// MI455X (gfx1250) — matrix operand order repaired and checked by reading, not compiled, not run on hardware
//
#include <hip/hip_runtime.h>

// ---------------- types for WMMA (gfx1250, wave32) ----------------
typedef __attribute__((ext_vector_type(16))) __bf16 v16bf;
typedef __attribute__((ext_vector_type(8)))  __bf16 v8bf;
typedef __attribute__((ext_vector_type(8)))  float  v8f;

// Problem constants (from the reference)
constexpr int kNTOKEN = 2048;
constexpr int kNINP   = 512;
constexpr int kNHID   = 1024;
constexpr int kB      = 64;
constexpr int kT      = 512;
constexpr int kKlow   = kNINP + kNHID + kNHID; // 2560 = emb|h_high|h_low
constexpr int kKhigh  = kNHID + kNHID;         // 2048 = h_low|h_high
constexpr long long kDecElems = (long long)kB * kT * kNTOKEN; // 67,108,864

constexpr int kRecBlocks  = 128;               // persistent grid: 1024 wave32s
constexpr int kRecThreads = 256;
constexpr int kPld = 17;                       // LDS partial row stride (pad vs 16)

__device__ __forceinline__ float sigf(float x) { return 1.0f / (1.0f + __expf(-x)); }

__device__ __forceinline__ v16bf cat16(v8bf lo, v8bf hi) {
  return __builtin_shufflevector(lo, hi, 0,1,2,3,4,5,6,7,8,9,10,11,12,13,14,15);
}

// One 16x16x32 WMMA: A fragment contiguous k = 16*half + s (two adjacent 16B chunks per lane).
__device__ __forceinline__ v8f wmma_step(const __bf16* abase, int rs, int mi,
                                         int idx16, int half, v16bf bm, v8f acc) {
  const __bf16* ar = abase + (size_t)((mi << 4) + idx16) * rs + half * 16;
  v8bf lo = *(const v8bf*)ar;
  v8bf hi = *(const v8bf*)(ar + 8);
  return __builtin_amdgcn_wmma_f32_16x16x32_bf16(false, cat16(lo, hi), false, bm,
                                                 (short)0, acc, false, false);
}

// C/D layout: VGPR r -> row (8*half + r), col n within the tile.
__device__ __forceinline__ void store_tile(float* base, int ldc, int m0, int n,
                                           int half, v8f a) {
  float* cr = base + (size_t)(m0 + half * 8) * ldc + n;
#pragma unroll
  for (int r = 0; r < 8; ++r) cr[(size_t)r * ldc] = a[r];
}

// Clean per-segment K loop: all bounds wave-uniform, pointers strength-reduce.
__device__ __forceinline__ void gemm_range(const __bf16* aseg, int rs,
                                           const __bf16* brow, int klo, int khi,
                                           int segbeg, int idx16, int half,
                                           v8f& a0, v8f& a1, v8f& a2, v8f& a3) {
  for (int k0 = klo; k0 < khi; k0 += 32) {
    const __bf16* abase = aseg + (k0 - segbeg);
    __builtin_prefetch(brow + k0 + 512, 0, 0);
    v16bf bm = *(const v16bf*)(brow + k0);
    a0 = wmma_step(abase, rs, 0, idx16, half, bm, a0);
    a1 = wmma_step(abase, rs, 1, idx16, half, bm, a1);
    a2 = wmma_step(abase, rs, 2, idx16, half, bm, a2);
    a3 = wmma_step(abase, rs, 3, idx16, half, bm, a3);
  }
}

// ---------------- weight conversion with gate-interleave permutation ----------------
// Output row p = j*4 + gate  (gate order i,f,g,o), cols = [W_ih | W_hh] bf16.
// A 16-wide n-tile then holds all 4 gates for 4 consecutive hidden units.
__global__ void k_build_wcat_perm(const float* __restrict__ Wih, const float* __restrict__ Whh,
                                  __bf16* __restrict__ out, int Kih, int Khh, long long total) {
  long long i = (long long)blockIdx.x * blockDim.x + threadIdx.x;
  long long stride = (long long)gridDim.x * blockDim.x;
  int K = Kih + Khh;
  for (; i < total; i += stride) {
    long long p = i / K;
    int c = (int)(i % K);
    int j = (int)(p >> 2), gate = (int)(p & 3);
    long long sr = (long long)gate * kNHID + j;
    float v = (c < Kih) ? Wih[sr * Kih + c] : Whh[sr * Khh + (c - Kih)];
    out[i] = (__bf16)v;
  }
}

__global__ void k_cvt(const float* __restrict__ src, __bf16* __restrict__ dst, long long n) {
  long long i = (long long)blockIdx.x * blockDim.x + threadIdx.x;
  long long stride = (long long)gridDim.x * blockDim.x;
  for (; i < n; i += stride) dst[i] = (__bf16)src[i];
}

// Pre-gather all embeddings: embAll[t][b][c] = emb[tokens[b,t]][c]  (bf16)
__global__ void k_gather_emb(const int* __restrict__ tokens, const float* __restrict__ emb,
                             __bf16* __restrict__ embAll) {
  long long total = (long long)kT * kB * kNINP;
  long long i = (long long)blockIdx.x * blockDim.x + threadIdx.x;
  long long stride = (long long)gridDim.x * blockDim.x;
  for (; i < total; i += stride) {
    int c = (int)(i % kNINP);
    long long tb = i / kNINP;
    int b = (int)(tb % kB);
    int t = (int)(tb / kB);
    embAll[i] = (__bf16)emb[(size_t)tokens[b * kT + t] * kNINP + c];
  }
}

// ---------------- init: seed xH from h0, zero grid barrier ----------------
__global__ void k_init(const float* __restrict__ h0l, const float* __restrict__ h0h,
                       __bf16* __restrict__ xH, unsigned* __restrict__ cnt) {
  int idx = blockIdx.x * blockDim.x + threadIdx.x;
  if (idx == 0) *cnt = 0u;
  if (idx >= kB * kNHID) return;
  int b = idx >> 10, j = idx & (kNHID - 1);
  xH[(size_t)b * kKhigh + j] = (__bf16)h0l[idx];
  xH[(size_t)b * kKhigh + kNHID + j] = (__bf16)h0h[idx];
}

// ---------------- grid-wide barrier ----------------
__device__ __forceinline__ void gbar(unsigned* cnt, int& epoch) {
  ++epoch;
  __threadfence();
  __syncthreads();
  if (threadIdx.x == 0) {
    atomicAdd(cnt, 1u);
    unsigned tgt = (unsigned)epoch * (unsigned)kRecBlocks;
    while (atomicAdd(cnt, 0u) < tgt) __builtin_amdgcn_s_sleep(1);
  }
  __syncthreads();
  __threadfence();
}

// split-K partial sum from LDS: waves tt*4+kc hold partials of n-tile tt
__device__ __forceinline__ float psum(const float* part, int tt, int b, int col) {
  const float* p = part + (size_t)(tt * 4) * (64 * kPld) + b * kPld + col;
  const int S = 64 * kPld;
  return (p[0] + p[S]) + (p[2 * S] + p[3 * S]);
}

// ---------------- persistent recurrent kernel: all 512 steps, 1 launch ----------------
// Block = 2 n-tiles x 4 split-K waves. Split-K reduction + LSTM nonlinearity stay
// inside the block (LDS + __syncthreads); only 2 grid barriers per step.
// Each thread keeps its c/h state and combined biases in registers for all steps.
__global__ void __launch_bounds__(kRecThreads)
k_recurrent(const int* __restrict__ tokens, const int* __restrict__ measure_p,
            const __bf16* __restrict__ embAll,
            const __bf16* __restrict__ WlowP, const __bf16* __restrict__ WhighP,
            const float* __restrict__ bih_l, const float* __restrict__ bhh_l,
            const float* __restrict__ bih_h, const float* __restrict__ bhh_h,
            const float* __restrict__ h0l, const float* __restrict__ c0l,
            const float* __restrict__ h0h, const float* __restrict__ c0h,
            __bf16* __restrict__ xH,
            float* __restrict__ hl, float* __restrict__ cl,
            float* __restrict__ hh, float* __restrict__ ch,
            __bf16* __restrict__ hsB, unsigned* __restrict__ cnt) {
  __shared__ float part[8 * 64 * kPld];            // 8 waves x 64 rows x 17 (~34 KB)
  const int bk   = blockIdx.x;
  const int wv   = threadIdx.x >> 5;               // 0..7
  const int lane = threadIdx.x & 31;
  const int half = lane >> 4, idx16 = lane & 15;
  const int ttl  = wv >> 2;                        // n-tile within block (0/1)
  const int kc   = wv & 3;                         // split-K chunk
  const int ncol = (((bk << 1) + ttl) << 4) + idx16;
  const int measure = measure_p[0];
  int epoch = 0;

  // GEMM pointers hoisted for the whole kernel
  const __bf16* browL = WlowP  + (size_t)ncol * kKlow  + half * 16;
  const __bf16* browH = WhighP + (size_t)ncol * kKhigh + half * 16;
  const int kbL = kc * (kKlow / 4),  keL = kbL + (kKlow / 4);
  const int kbH = kc * (kKhigh / 4), keH = kbH + (kKhigh / 4);

  // elementwise ownership: hidden unit j (fixed), batches b and b+32
  const int jl = threadIdx.x & 7;                  // 0..7 within block
  const int bb0 = threadIdx.x >> 3;                // 0..31
  const int j = (bk << 3) + jl;                    // global hidden unit
  const int tt = jl >> 2, cb = (jl & 3) << 2;      // LDS tile / column base
  float Bl[4], Bh[4];
#pragma unroll
  for (int g = 0; g < 4; ++g) {
    Bl[g] = bih_l[g * kNHID + j] + bhh_l[g * kNHID + j];
    Bh[g] = bih_h[g * kNHID + j] + bhh_h[g * kNHID + j];
  }
  float cL[2], cH[2], hL[2], hH[2];
#pragma unroll
  for (int oi = 0; oi < 2; ++oi) {
    int b = bb0 + oi * 32;
    cL[oi] = c0l[b * kNHID + j];
    cH[oi] = c0h[b * kNHID + j];
    hL[oi] = h0l[b * kNHID + j];
    hH[oi] = h0h[b * kNHID + j];
  }

  for (int t = 0; t < kT; ++t) {
    // ---- G1: low-cell gates, A = [emb(t) | h_high | h_low], K = 2560 ----
    {
      v8f a0 = {}, a1 = {}, a2 = {}, a3 = {};
      const __bf16* embT = embAll + (size_t)t * kB * kNINP;
      int lo = kbL, hi = keL < kNINP ? keL : kNINP;
      if (lo < hi) gemm_range(embT, kNINP, browL, lo, hi, 0, idx16, half, a0, a1, a2, a3);
      lo = kbL > kNINP ? kbL : kNINP;
      hi = keL < kNINP + kNHID ? keL : kNINP + kNHID;
      if (lo < hi) gemm_range(xH + kNHID, kKhigh, browL, lo, hi, kNINP, idx16, half,
                              a0, a1, a2, a3);
      lo = kbL > kNINP + kNHID ? kbL : kNINP + kNHID;
      hi = keL;
      if (lo < hi) gemm_range(xH, kKhigh, browL, lo, hi, kNINP + kNHID, idx16, half,
                              a0, a1, a2, a3);
      float* pp = part + (size_t)wv * (64 * kPld);
      store_tile(pp, kPld, 0,  idx16, half, a0);
      store_tile(pp, kPld, 16, idx16, half, a1);
      store_tile(pp, kPld, 32, idx16, half, a2);
      store_tile(pp, kPld, 48, idx16, half, a3);
    }
    __syncthreads();
    // ---- E1: low-cell LSTM update from LDS partials (deterministic sum order) ----
#pragma unroll
    for (int oi = 0; oi < 2; ++oi) {
      int b = bb0 + oi * 32;
      float gi = psum(part, tt, b, cb + 0) + Bl[0];
      float gf = psum(part, tt, b, cb + 1) + Bl[1];
      float gg = psum(part, tt, b, cb + 2) + Bl[2];
      float go = psum(part, tt, b, cb + 3) + Bl[3];
      float cn = sigf(gf) * cL[oi] + sigf(gi) * tanhf(gg);
      float hn = sigf(go) * tanhf(cn);
      cL[oi] = cn;
      hL[oi] = hn;
      __bf16 hb = (__bf16)hn;
      xH[(size_t)b * kKhigh + j] = hb;                       // h_low half of xH
      hsB[(size_t)t * kB * kNHID + (size_t)b * kNHID + j] = hb;
    }
    gbar(cnt, epoch);                                        // h_low visible device-wide

    // ---- G2: high-cell gates, A = xH = [h_low | h_high], K = 2048 ----
    {
      v8f a0 = {}, a1 = {}, a2 = {}, a3 = {};
      gemm_range(xH, kKhigh, browH, kbH, keH, 0, idx16, half, a0, a1, a2, a3);
      float* pp = part + (size_t)wv * (64 * kPld);
      store_tile(pp, kPld, 0,  idx16, half, a0);
      store_tile(pp, kPld, 16, idx16, half, a1);
      store_tile(pp, kPld, 32, idx16, half, a2);
      store_tile(pp, kPld, 48, idx16, half, a3);
    }
    __syncthreads();
    // ---- E2: high-cell update, gated by (token == measure) ----
#pragma unroll
    for (int oi = 0; oi < 2; ++oi) {
      int b = bb0 + oi * 32;
      float filt = (tokens[b * kT + t] == measure) ? 1.0f : 0.0f;
      float gi = psum(part, tt, b, cb + 0) + Bh[0];
      float gf = psum(part, tt, b, cb + 1) + Bh[1];
      float gg = psum(part, tt, b, cb + 2) + Bh[2];
      float go = psum(part, tt, b, cb + 3) + Bh[3];
      float cn = sigf(gf) * cH[oi] + sigf(gi) * tanhf(gg);
      float hn = sigf(go) * tanhf(cn);
      float hnew = filt * hn + (1.0f - filt) * hH[oi];
      float cnew = filt * cn + (1.0f - filt) * cH[oi];
      hH[oi] = hnew;
      cH[oi] = cnew;
      xH[(size_t)b * kKhigh + kNHID + j] = (__bf16)hnew;     // h_high half of xH
    }
    gbar(cnt, epoch);                                        // h_high visible device-wide
  }

  // final states from registers
#pragma unroll
  for (int oi = 0; oi < 2; ++oi) {
    int b = bb0 + oi * 32;
    hl[b * kNHID + j] = hL[oi];
    cl[b * kNHID + j] = cL[oi];
    hh[b * kNHID + j] = hH[oi];
    ch[b * kNHID + j] = cH[oi];
  }
}

// ---------------- decode: out[b,t,n] = hs[t,b,:].Wdec[n,:] + bdec[n] ----------------
// Block = 64 rows x 128 cols; A chunks (64x256 bf16) double-buffered in LDS via
// global_load_async_to_lds_b128 (ASYNCcnt) overlapping the WMMA of the live chunk.
constexpr int kArs = 264;          // LDS row stride: 528B, 16B-aligned, bank-skewed
__global__ void __launch_bounds__(256)
k_decode(const __bf16* __restrict__ hs, const __bf16* __restrict__ Wdec,
         const float* __restrict__ bdec, float* __restrict__ out) {
  __shared__ __bf16 As[2][64 * kArs];              // ~66 KB, double-buffered
  const int mg = blockIdx.x >> 4;                  // 0..511 : 64-row group
  const int ng = blockIdx.x & 15;                  // 0..15  : 128-col group
  const int m0 = mg << 6;
  const int bb = m0 >> 9;                          // batch index (64 | 512)
  const int t0 = m0 & (kT - 1);
  const int wv = threadIdx.x >> 5, lane = threadIdx.x & 31;
  const int half = lane >> 4, idx16 = lane & 15;
  const int n0 = (ng << 7) + (wv << 4);
  const __bf16* brow = Wdec + (size_t)(n0 + idx16) * kNHID + half * 16;
  v8f acc0 = {}, acc1 = {}, acc2 = {}, acc3 = {};

  auto issue = [&](int kcb, int buf) {
    for (int v = threadIdx.x; v < 64 * 32; v += 256) {
      int r = v >> 5, cv = (v & 31) << 3;
      const __bf16* g = hs + ((size_t)(t0 + r) * kB + bb) * kNHID + kcb + cv;
      unsigned ldso = (unsigned)(size_t)(&As[buf][r * kArs + cv]);
      asm volatile("global_load_async_to_lds_b128 %0, %1, off"
                   :: "v"(ldso), "v"(g) : "memory");
    }
  };

  issue(0, 0);
  for (int c = 0; c < 4; ++c) {
    asm volatile("s_wait_asynccnt 0x0" ::: "memory");   // chunk c landed in LDS
    __syncthreads();                                    // all waves' loads + buffer free
    if (c + 1 < 4) issue((c + 1) << 8, (c + 1) & 1);    // prefetch next chunk async
    const __bf16* ab = As[c & 1];
    const int kcb = c << 8;
    for (int kk = 0; kk < 256; kk += 32) {
      __builtin_prefetch(brow + kcb + kk + 512, 0, 0);
      v16bf bm = *(const v16bf*)(brow + kcb + kk);
      acc0 = wmma_step(ab + kk, kArs, 0, idx16, half, bm, acc0);
      acc1 = wmma_step(ab + kk, kArs, 1, idx16, half, bm, acc1);
      acc2 = wmma_step(ab + kk, kArs, 2, idx16, half, bm, acc2);
      acc3 = wmma_step(ab + kk, kArs, 3, idx16, half, bm, acc3);
    }
  }
  int n = n0 + idx16;
  float bias = bdec[n];
  v8f accs[4] = {acc0, acc1, acc2, acc3};
#pragma unroll
  for (int mi = 0; mi < 4; ++mi) {
    float* cr = out + (size_t)(m0 + (mi << 4) + half * 8) * kNTOKEN + n;
#pragma unroll
    for (int r = 0; r < 8; ++r) cr[(size_t)r * kNTOKEN] = accs[mi][r] + bias;
  }
}

// ---------------- append final states to d_out ----------------
__global__ void k_final(const float* __restrict__ hl, const float* __restrict__ cl,
                        const float* __restrict__ hh, const float* __restrict__ ch,
                        float* __restrict__ out) {
  int idx = blockIdx.x * blockDim.x + threadIdx.x;
  int n = kB * kNHID;
  if (idx >= 4 * n) return;
  int which = idx / n, jj = idx % n;
  float v = (which == 0) ? hl[jj] : (which == 1) ? cl[jj] : (which == 2) ? hh[jj] : ch[jj];
  out[idx] = v;
}

// ---------------- host orchestration: 8 launches total ----------------
extern "C" void kernel_launch(void* const* d_in, const int* in_sizes, int n_in,
                              void* d_out, int out_size, void* d_ws, size_t ws_size,
                              hipStream_t stream) {
  (void)in_sizes; (void)n_in; (void)out_size; (void)ws_size;
  const int*   tokens  = (const int*)d_in[0];
  const int*   measure = (const int*)d_in[1];
  const float* emb     = (const float*)d_in[2];
  const float* Wih_l   = (const float*)d_in[3];
  const float* Whh_l   = (const float*)d_in[4];
  const float* bih_l   = (const float*)d_in[5];
  const float* bhh_l   = (const float*)d_in[6];
  const float* Wih_h   = (const float*)d_in[7];
  const float* Whh_h   = (const float*)d_in[8];
  const float* bih_h   = (const float*)d_in[9];
  const float* bhh_h   = (const float*)d_in[10];
  const float* Wdec    = (const float*)d_in[11];
  const float* bdec    = (const float*)d_in[12];
  const float* h0l     = (const float*)d_in[13];
  const float* c0l     = (const float*)d_in[14];
  const float* h0h     = (const float*)d_in[15];
  const float* c0h     = (const float*)d_in[16];
  float* out = (float*)d_out;

  char* wsp = (char*)d_ws;
  size_t off = 0;
  auto carve = [&](size_t bytes) -> void* {
    void* p = wsp + off;
    off += (bytes + 255) & ~(size_t)255;
    return p;
  };
  __bf16* WlowP  = (__bf16*)carve((size_t)4 * kNHID * kKlow * 2);  // 20 MB, gate-permuted
  __bf16* WhighP = (__bf16*)carve((size_t)4 * kNHID * kKhigh * 2); // 16 MB, gate-permuted
  __bf16* WdecB  = (__bf16*)carve((size_t)kNTOKEN * kNHID * 2);    //  4 MB
  __bf16* embAll = (__bf16*)carve((size_t)kT * kB * kNINP * 2);    // 32 MB
  __bf16* hsB    = (__bf16*)carve((size_t)kT * kB * kNHID * 2);    // 64 MB
  __bf16* xH     = (__bf16*)carve((size_t)kB * kKhigh * 2);
  float*  hl     = (float*)carve((size_t)kB * kNHID * 4);
  float*  cl     = (float*)carve((size_t)kB * kNHID * 4);
  float*  hh     = (float*)carve((size_t)kB * kNHID * 4);
  float*  ch     = (float*)carve((size_t)kB * kNHID * 4);
  unsigned* cnt  = (unsigned*)carve(256);

  k_build_wcat_perm<<<2048, 256, 0, stream>>>(Wih_l, Whh_l, WlowP, kNINP + kNHID, kNHID,
                                              (long long)4 * kNHID * kKlow);
  k_build_wcat_perm<<<2048, 256, 0, stream>>>(Wih_h, Whh_h, WhighP, kNHID, kNHID,
                                              (long long)4 * kNHID * kKhigh);
  k_cvt<<<1024, 256, 0, stream>>>(Wdec, WdecB, (long long)kNTOKEN * kNHID);
  k_gather_emb<<<4096, 256, 0, stream>>>(tokens, emb, embAll);
  k_init<<<256, 256, 0, stream>>>(h0l, h0h, xH, cnt);

  k_recurrent<<<kRecBlocks, kRecThreads, 0, stream>>>(
      tokens, measure, embAll, WlowP, WhighP, bih_l, bhh_l, bih_h, bhh_h,
      h0l, c0l, h0h, c0h, xH, hl, cl, hh, ch, hsB, cnt);

  k_decode<<<8192, 256, 0, stream>>>(hsB, WdecB, bdec, out);
  k_final<<<1024, 256, 0, stream>>>(hl, cl, hh, ch, out + kDecElems);
}
